// DeltaGRUModel_32873679684043
// MI455X (gfx1250) — compile-verified, matrix operand order checked by reading, not run on hardware
//
#include <hip/hip_runtime.h>

#define HDIM   256
#define TDIM   4096
#define BDIM   32
#define FIN    6
#define TH_X   0.05f
#define TH_H   0.05f

typedef __bf16 bf16_t;
typedef __attribute__((ext_vector_type(16))) __bf16 v16bf;
typedef __attribute__((ext_vector_type(8)))  float  v8f;

// ---------------- workspace layout (bytes) ----------------
#define WS_FEATS 0
#define WS_SKIP  3145728
#define WS_WHP   4194304

// dynamic LDS layout for GRU kernel (bytes)
//   Wrz  : 262144  (r/z gate fragments, 32 tiles)
//   dhf  : 16384   (double-buffered A-fragments, 2 x 8KB)
//   WxL  : 18432
//   dxs  : 384
//   oacc : 256     (double-buffered 2 x 32 f32)
#define LDS_WRZ   0
#define LDS_DHF   262144
#define LDS_WX    (262144 + 16384)
#define LDS_DXS   (262144 + 16384 + 18432)
#define LDS_OACC  (262144 + 16384 + 18432 + 384)
#define LDS_TOTAL (262144 + 16384 + 18432 + 384 + 256)

__device__ __forceinline__ float hswish(float v) {
    return v * fminf(fmaxf(v + 3.0f, 0.0f), 6.0f) * (1.0f / 6.0f);
}
__device__ __forceinline__ float fsigmoid(float x) {
    return 1.0f / (1.0f + __expf(-x));
}
__device__ __forceinline__ float ftanh(float x) {
    float e2 = __expf(2.0f * x);
    return 1.0f - 2.0f / (e2 + 1.0f);
}

// gfx1250 async global->LDS copy (ASYNCcnt-tracked)
__device__ __forceinline__ void async_ld_lds_b128(void* lds_dst, const void* gsrc) {
    unsigned ldsoff = (unsigned)(uintptr_t)lds_dst;          // low 32 bits = LDS offset
    unsigned long long ga = (unsigned long long)(uintptr_t)gsrc;
    asm volatile("global_load_async_to_lds_b128 %0, %1, off"
                 :: "v"(ldsoff), "v"(ga) : "memory");
}
__device__ __forceinline__ void wait_asynccnt0() {
    asm volatile("s_wait_asynccnt 0x0" ::: "memory");
}

// --------------------------------------------------------------------------
// Pack Wh (f32, [768][256]) into 16x16x32 bf16 B-fragment layout.
// --------------------------------------------------------------------------
__global__ void dgru_pack_wh(const float* __restrict__ Wh, bf16_t* __restrict__ whp) {
    int idx = blockIdx.x * 256 + threadIdx.x;
    if (idx >= 768 * 256) return;
    int n = idx >> 8;
    int k = idx & 255;
    int nt = n >> 4, nn = n & 15;
    int kt = k >> 5, kk = k & 31;
    int half = (kk >> 3) & 1;
    int s = (kk & 7) + 8 * (kk >> 4);
    int lane = nn + 16 * half;
    whp[(((nt * 8 + kt) * 32 + lane) << 4) + s] = (bf16_t)Wh[idx];
}

// --------------------------------------------------------------------------
// Prep: feats = [i, q, amp, amp^3, i_prev, q_prev] and conv skip path.
// --------------------------------------------------------------------------
__global__ void dgru_prep(const float* __restrict__ x,
                          const float* __restrict__ Wc1,   // (3,2,3) OIH
                          const float* __restrict__ Wc2,   // (2,3,1)
                          float* __restrict__ feats,
                          float* __restrict__ skip) {
    int idx = blockIdx.x * 256 + threadIdx.x;
    if (idx >= BDIM * TDIM) return;
    int b = idx / TDIM, t = idx % TDIM;
    const float* xb = x + (size_t)b * TDIM * 2;

    float i = xb[t * 2 + 0], q = xb[t * 2 + 1];
    float amp2 = i * i + q * q;
    float amp = sqrtf(fmaxf(amp2, 1e-12f));
    float amp3 = amp * amp2;
    float ip = (t > 0) ? xb[(t - 1) * 2 + 0] : 0.0f;
    float qp = (t > 0) ? xb[(t - 1) * 2 + 1] : 0.0f;
    float* f = feats + (size_t)idx * FIN;
    f[0] = i; f[1] = q; f[2] = amp; f[3] = amp3; f[4] = ip; f[5] = qp;

    float c1[3];
#pragma unroll
    for (int o = 0; o < 3; ++o) {
        float acc = 0.0f;
#pragma unroll
        for (int kk = 0; kk < 3; ++kk) {
            int tt = t + 16 * (kk - 1);
            if (tt >= 0 && tt < TDIM) {
                acc += xb[tt * 2 + 0] * Wc1[(o * 2 + 0) * 3 + kk]
                     + xb[tt * 2 + 1] * Wc1[(o * 2 + 1) * 3 + kk];
            }
        }
        c1[o] = hswish(acc);
    }
#pragma unroll
    for (int o = 0; o < 2; ++o) {
        float acc = c1[0] * Wc2[o * 3 + 0] + c1[1] * Wc2[o * 3 + 1] + c1[2] * Wc2[o * 3 + 2];
        skip[(size_t)idx * 2 + o] = hswish(acc);
    }
}

// --------------------------------------------------------------------------
// Persistent Delta-GRU scan: 2 blocks x 512 threads (16 waves).
// Wave w owns hidden columns [16w,16w+16) of all three gate bands.
// --------------------------------------------------------------------------
__global__ __launch_bounds__(512) void dgru_scan(
        const float* __restrict__ feats,
        const float* __restrict__ skip,
        const float* __restrict__ Wx,    // (768,6) f32
        const bf16_t* __restrict__ whp,  // packed Wh fragments
        const float* __restrict__ Wo,    // (2,256) f32
        float* __restrict__ out) {       // (B,T,2) f32
    extern __shared__ char smem[];
    bf16_t* Wrz = (bf16_t*)(smem + LDS_WRZ);
    bf16_t* dhf = (bf16_t*)(smem + LDS_DHF);     // 2 buffers x 4096 elems
    float*  WxL = (float*)(smem + LDS_WX);
    float*  dxs = (float*)(smem + LDS_DXS);
    float*  oacc = (float*)(smem + LDS_OACC);    // 2 buffers x 32 f32

    const int tid = threadIdx.x;
    const int w = tid >> 5;
    const int L = tid & 31;
    const int ln = L & 15;
    const int hi = L >> 4;
    const int b0 = blockIdx.x * 16;
    const int j = w * 16 + ln;

    // ---- one-time staging: r/z fragments via async global->LDS DMA ----
    for (int i = tid; i < 16384; i += 512) {
        async_ld_lds_b128((char*)Wrz + (size_t)i * 16, (const char*)whp + (size_t)i * 16);
    }
    for (int i = tid; i < 768 * 6; i += 512) WxL[i] = Wx[i];
    {
        unsigned* dz = (unsigned*)dhf;
        for (int i = tid; i < 4096; i += 512) dz[i] = 0u;   // both dhf buffers
    }
    if (tid < 64) oacc[tid] = 0.0f;                          // both oacc buffers
    wait_asynccnt0();
    __syncthreads();

    // ---- per-lane recurrent state registers: rows m = v + 8*hi, col j ----
    float hreg[8], hp[8], dmnh[8], dmr[8], dmz[8], dmn[8];
#pragma unroll
    for (int v = 0; v < 8; ++v) {
        hreg[v] = hp[v] = dmnh[v] = dmr[v] = dmz[v] = dmn[v] = 0.0f;
    }
    const float wo0 = Wo[j];
    const float wo1 = Wo[256 + j];

    const bf16_t* whn = whp + (size_t)((32 + w) * 8) * 512;  // n-gate tiles for this wave
    const int m16 = tid / 6, ff = tid % 6;                   // delta-x ownership (tid<96)
    float xp = 0.0f;

    const int kt2 = j >> 5, kk0 = j & 31;
    const int half2 = (kk0 >> 3) & 1;
    const int s2 = (kk0 & 7) + 8 * (kk0 >> 4);

    for (int t = 0; t < TDIM; ++t) {
        const int cur = t & 1, nxt = (t + 1) & 1;
        bf16_t* dhfR = dhf + cur * 4096;    // read by WMMA this step
        bf16_t* dhfW = dhf + nxt * 4096;    // written by gate update this step

        // n-gate B fragments: stream from L2 every step. Launder the pointer
        // (VGPR constraint: address is wave-divergent) so the loads cannot be
        // hoisted out of the loop and spilled; issue before the barrier so
        // L2 latency overlaps it.
        uintptr_t bl = (uintptr_t)whn;
        asm volatile("" : "+v"(bl));
        const bf16_t* bnl = (const bf16_t*)bl;
        v16bf bnv[8];
#pragma unroll
        for (int kt = 0; kt < 8; ++kt) {
            bnv[kt] = *(const v16bf*)(bnl + kt * 512 + L * 16);
        }

        // (a) delta-x
        if (tid < 96) {
            const float* fp = &feats[((size_t)(b0 + m16) * TDIM + t) * FIN + ff];
            float xt = *fp;
            __builtin_prefetch(fp + 8 * FIN, 0, 0);
            float dx = xt - xp;
            float a = fabsf(dx);
            dxs[m16 * 6 + ff] = (a < TH_X) ? 0.0f : dx;
            if (a >= TH_X) xp = xt;
        }
        __syncthreads();   // dhfR (prev step's writes) + dxs ready

        // (d) phase A: dm += dx @ Wx^T for owned (m, j)
#pragma unroll
        for (int f = 0; f < 6; ++f) {
            float wr = WxL[j * 6 + f];
            float wz = WxL[(j + 256) * 6 + f];
            float wn = WxL[(j + 512) * 6 + f];
#pragma unroll
            for (int v = 0; v < 8; ++v) {
                float d = dxs[(v + 8 * hi) * 6 + f];
                dmr[v] += d * wr; dmz[v] += d * wz; dmn[v] += d * wn;
            }
        }
        if (tid < 32) oacc[nxt * 32 + tid] = 0.0f;   // zero NEXT step's buffer

        // (c) mac_h = dh @ Wh^T via WMMA bf16: 3 gate tiles x 8 k-steps
        v8f cr = {}, cz = {}, cn = {};
#pragma unroll
        for (int kt = 0; kt < 8; ++kt) {
            v16bf a  = *(const v16bf*)(dhfR + kt * 512 + L * 16);
            v16bf br = *(const v16bf*)(Wrz + (w * 8 + kt) * 512 + L * 16);
            v16bf bz = *(const v16bf*)(Wrz + ((16 + w) * 8 + kt) * 512 + L * 16);
            cr = __builtin_amdgcn_wmma_f32_16x16x32_bf16(false, a, false, br, (short)0, cr, false, false);
            cz = __builtin_amdgcn_wmma_f32_16x16x32_bf16(false, a, false, bz, (short)0, cz, false, false);
            cn = __builtin_amdgcn_wmma_f32_16x16x32_bf16(false, a, false, bnv[kt], (short)0, cn, false, false);
        }

        // (e) gate math + h update; dh fragments go to the OTHER buffer,
        // so no barrier is needed between WMMA reads and these writes.
#pragma unroll
        for (int v = 0; v < 8; ++v) {
            int m = v + 8 * hi;
            float mr = dmr[v] + cr[v];
            float mz = dmz[v] + cz[v];
            dmr[v] = mr; dmz[v] = mz;
            dmnh[v] += cn[v];
            float r = fsigmoid(mr);
            float z = fsigmoid(mz);
            float n = ftanh(dmn[v] + r * dmnh[v]);
            float hn = (1.0f - z) * n + z * hreg[v];
            hreg[v] = hn;
            float dh = hn - hp[v];
            float ad = fabsf(dh);
            float dhv = (ad < TH_H) ? 0.0f : dh;
            if (ad >= TH_H) hp[v] = hn;
            dhfW[kt2 * 512 + (m + 16 * half2) * 16 + s2] = (bf16_t)dhv;
            atomicAdd(&oacc[cur * 32 + m * 2 + 0], hn * wo0);
            atomicAdd(&oacc[cur * 32 + m * 2 + 1], hn * wo1);
        }
        __syncthreads();   // oacc[cur] complete (also orders dhfW for next step)

        if (tid < 32) {
            int m = tid >> 1, o = tid & 1;
            size_t oi = ((size_t)(b0 + m) * TDIM + t) * 2 + o;
            out[oi] = oacc[cur * 32 + tid] + skip[oi];
        }
    }
}

extern "C" void kernel_launch(void* const* d_in, const int* in_sizes, int n_in,
                              void* d_out, int out_size, void* d_ws, size_t ws_size,
                              hipStream_t stream) {
    const float* x   = (const float*)d_in[0];
    const float* Wx  = (const float*)d_in[1];
    const float* Wh  = (const float*)d_in[2];
    const float* Wo  = (const float*)d_in[3];
    const float* Wc1 = (const float*)d_in[4];
    const float* Wc2 = (const float*)d_in[5];
    float* out = (float*)d_out;

    char* ws = (char*)d_ws;
    float*  feats = (float*)(ws + WS_FEATS);
    float*  skipb = (float*)(ws + WS_SKIP);
    bf16_t* whp   = (bf16_t*)(ws + WS_WHP);

    (void)hipFuncSetAttribute(reinterpret_cast<const void*>(dgru_scan),
                              hipFuncAttributeMaxDynamicSharedMemorySize,
                              (int)LDS_TOTAL);

    dgru_pack_wh<<<768, 256, 0, stream>>>(Wh, whp);
    dgru_prep<<<(BDIM * TDIM + 255) / 256, 256, 0, stream>>>(x, Wc1, Wc2, feats, skipb);
    dgru_scan<<<2, 512, LDS_TOTAL, stream>>>(feats, skipb, Wx, whp, Wo, out);
}
